// IOUfitTransformer_75831942578676
// MI455X (gfx1250) — matrix operand order repaired and checked by reading, not compiled, not run on hardware
//
#include <hip/hip_runtime.h>
#include <hip/hip_bf16.h>
#include <hip/hip_fp16.h>

// ---------------------------------------------------------------------------
// Types matching CDNA5 WMMA operand shapes (wave32)
// ---------------------------------------------------------------------------
typedef _Float16 half16 __attribute__((ext_vector_type(16)));
typedef _Float16 half8  __attribute__((ext_vector_type(8)));
typedef float    f32x8  __attribute__((ext_vector_type(8)));
typedef float    f32x4  __attribute__((ext_vector_type(4)));

constexpr int BSZ  = 32;
constexpr int NSEQ = 2048;
constexpr int CDIM = 512;

__device__ __forceinline__ f32x8 zero8() { f32x8 z = {0.f,0.f,0.f,0.f,0.f,0.f,0.f,0.f}; return z; }

__device__ __forceinline__ half16 cat16(half8 lo, half8 hi) {
  half16 r;
#pragma unroll
  for (int i = 0; i < 8; ++i) { r[i] = lo[i]; r[i + 8] = hi[i]; }
  return r;
}

// D = A(16x32 f16) * B(32x16 f16) + C(16x16 f32)
__device__ __forceinline__ f32x8 wmma_f16(half16 a, half16 b, f32x8 c) {
  return __builtin_amdgcn_wmma_f32_16x16x32_f16(false, a, false, b, (short)0, c, false, false);
}

// A-fragment (16x32, 16-bit): lane<16 -> row=lane,   K = {k0..k0+7,  k0+16..k0+23}
//                             lane>=16 -> row=lane-16, K = {k0+8..15, k0+24..31}
__device__ __forceinline__ half16 load_a_f16(const _Float16* base, int ldr, int row,
                                             int k0, int lane) {
  const _Float16* p = base + (size_t)row * ldr + k0 + ((lane >= 16) ? 8 : 0);
  half8 lo = *(const half8*)p;
  half8 hi = *(const half8*)(p + 16);
  return cat16(lo, hi);
}

__device__ __forceinline__ half16 load_a_f32(const float* base, int ldr, int row,
                                             int k0, int lane) {
  const float* p = base + (size_t)row * ldr + k0 + ((lane >= 16) ? 8 : 0);
  f32x4 a0 = *(const f32x4*)(p);
  f32x4 a1 = *(const f32x4*)(p + 4);
  f32x4 a2 = *(const f32x4*)(p + 16);
  f32x4 a3 = *(const f32x4*)(p + 20);
  half16 r;
#pragma unroll
  for (int i = 0; i < 4; ++i) {
    r[i]      = (_Float16)a0[i];
    r[i + 4]  = (_Float16)a1[i];
    r[i + 8]  = (_Float16)a2[i];
    r[i + 12] = (_Float16)a3[i];
  }
  return r;
}

// B-fragment (32x16, 16-bit): lane<16 -> col=lane,    K = {k0..k0+7,  k0+16..k0+23}
//                             lane>=16 -> col=lane-16, K = {k0+8..15, k0+24..31}
// colbase must already point at this lane's column (column-contiguous storage).
__device__ __forceinline__ half16 load_b_f16(const _Float16* colbase, int k0, int lane) {
  const _Float16* p = colbase + k0 + ((lane >= 16) ? 8 : 0);
  half8 lo = *(const half8*)p;
  half8 hi = *(const half8*)(p + 16);
  return cat16(lo, hi);
}

// ---------------------------------------------------------------------------
// Kernel 0: f32 [K,N] weights -> f16 transposed [N,K] (contiguous B columns)
// ---------------------------------------------------------------------------
__global__ void prep_weights(const float* __restrict__ wq, const float* __restrict__ wk,
                             const float* __restrict__ wv, const float* __restrict__ wo,
                             _Float16* __restrict__ wqT, _Float16* __restrict__ wkT,
                             _Float16* __restrict__ wvT, _Float16* __restrict__ woT) {
  int i = blockIdx.x * blockDim.x + threadIdx.x;   // i = k*512 + n
  int k = i >> 9, n = i & 511;
  int t = n * CDIM + k;
  wqT[t] = (_Float16)wq[i];
  wkT[t] = (_Float16)wk[i];
  wvT[t] = (_Float16)wv[i];
  woT[t] = (_Float16)wo[i];
}

// ---------------------------------------------------------------------------
// Kernel 1: Y = X(f32)*Wt' + b  -> f16 (optionally stored transposed [B,D,N])
// One wave computes a 16x64 tile. Grid: 8192 blocks x 128 threads (4 waves).
// ---------------------------------------------------------------------------
__global__ void proj_f32(const float* __restrict__ X, const _Float16* __restrict__ Wt,
                         const float* __restrict__ bias, _Float16* __restrict__ Y,
                         int transposeV) {
  int lane = threadIdx.x & 31;
  int wid  = blockIdx.x * 4 + (threadIdx.x >> 5);   // 0..32767
  int rt = wid >> 3, cg = wid & 7;
  int mbase = rt * 16, n0 = cg * 64;
  int lr = lane & 15;

  f32x8 acc[4];
#pragma unroll
  for (int t = 0; t < 4; ++t) acc[t] = zero8();

#pragma unroll
  for (int s = 0; s < 16; ++s) {
    int k0 = 32 * s;
    half16 a = load_a_f32(X, CDIM, mbase + lr, k0, lane);
#pragma unroll
    for (int t = 0; t < 4; ++t) {
      half16 b = load_b_f16(Wt + (size_t)(n0 + t * 16 + lr) * CDIM, k0, lane);
      acc[t] = wmma_f16(a, b, acc[t]);
    }
  }

  int hi = (lane >= 16) ? 8 : 0;
#pragma unroll
  for (int t = 0; t < 4; ++t) {
    int col = n0 + t * 16 + lr;
    float bv = bias[col];
#pragma unroll
    for (int r = 0; r < 8; ++r) {
      int row = mbase + r + hi;                       // global row in [0, B*N)
      float v = acc[t][r] + bv;
      if (!transposeV) {
        Y[(size_t)row * CDIM + col] = (_Float16)v;
      } else {
        int bidx = row >> 11;                         // N = 2048
        int nr   = row & (NSEQ - 1);
        Y[((size_t)bidx * CDIM + col) * NSEQ + nr] = (_Float16)v;
      }
    }
  }
}

// ---------------------------------------------------------------------------
// Kernel 2: flash attention.  One wave per (batch, 16-query tile).
// scores = QK^T / 512, online softmax, O = P*V, final scale 1/(l * 2048).
// __launch_bounds__(128, 1): allow the full >256-VGPR budget (gfx1250 MSB
// addressing) so the 16 resident Q fragments + 32 f32 accumulators do not
// spill to scratch.
// ---------------------------------------------------------------------------
__global__ void __launch_bounds__(128, 1)
flash_attn(const _Float16* __restrict__ Qm, const _Float16* __restrict__ Km,
           const _Float16* __restrict__ Vt, _Float16* __restrict__ Out) {
  __shared__ _Float16 smem[4][16][40];               // per-wave 16x32 P tile (+pad)
  int lane = threadIdx.x & 31;
  int widx = threadIdx.x >> 5;
  int wid  = blockIdx.x * 4 + widx;                  // 0..4095
  int b = wid >> 7, qt = wid & 127;

  const _Float16* qb = Qm + (size_t)b * NSEQ * CDIM;
  const _Float16* kb = Km + (size_t)b * NSEQ * CDIM;
  const _Float16* vb = Vt + (size_t)b * CDIM * NSEQ;

  int lr = lane & 15;
  int hi = (lane >= 16) ? 8 : 0;
  int qrow = qt * 16 + lr;

  // Resident Q (16 A-fragments = full 16x512 tile)
  half16 qf[16];
#pragma unroll
  for (int s = 0; s < 16; ++s) qf[s] = load_a_f16(qb, CDIM, qrow, 32 * s, lane);

  f32x8 o[32];
#pragma unroll
  for (int dt = 0; dt < 32; ++dt) o[dt] = zero8();

  float m[8], l[8];
#pragma unroll
  for (int r = 0; r < 8; ++r) { m[r] = -1.0e30f; l[r] = 0.0f; }

  const float inv_d = 1.0f / 512.0f;

  for (int kt = 0; kt < 64; ++kt) {                  // 32 keys per chunk
    int kbase = kt * 32;

    // Prefetch next chunk's K rows into cache while this chunk computes
    // (lowers to global_prefetch_b8; no wait counters involved).
    if (kt + 1 < 64) {
      const _Float16* nk = kb + (size_t)(kbase + 32 + lane) * CDIM;
      __builtin_prefetch(nk, 0, 3);
      __builtin_prefetch(nk + 256, 0, 3);
    }

    f32x8 s0 = zero8(), s1 = zero8();

#pragma unroll
    for (int s = 0; s < 16; ++s) {
      int k0 = 32 * s;
      half16 b0 = load_b_f16(kb + (size_t)(kbase + lr) * CDIM, k0, lane);
      half16 b1 = load_b_f16(kb + (size_t)(kbase + 16 + lr) * CDIM, k0, lane);
      s0 = wmma_f16(qf[s], b0, s0);
      s1 = wmma_f16(qf[s], b1, s1);
    }

    // Online softmax.  Accumulator component r belongs to row (r + hi); the
    // 16-lane shfl_xor reductions stay inside each half-wave group, which is
    // exactly the set of lanes holding that row.
    float scale[8];
#pragma unroll
    for (int r = 0; r < 8; ++r) {
      float v0 = s0[r] * inv_d;
      float v1 = s1[r] * inv_d;
      float mx = fmaxf(v0, v1);
#pragma unroll
      for (int off = 1; off < 16; off <<= 1) mx = fmaxf(mx, __shfl_xor(mx, off, 32));
      float mn = fmaxf(m[r], mx);
      float sc = __expf(m[r] - mn);
      float p0 = __expf(v0 - mn);
      float p1 = __expf(v1 - mn);
      float sum = p0 + p1;
#pragma unroll
      for (int off = 1; off < 16; off <<= 1) sum += __shfl_xor(sum, off, 32);
      l[r] = l[r] * sc + sum;
      m[r] = mn;
      scale[r] = sc;
      s0[r] = p0;
      s1[r] = p1;
    }

#pragma unroll
    for (int dt = 0; dt < 32; ++dt)
#pragma unroll
      for (int r = 0; r < 8; ++r) o[dt][r] *= scale[r];

    // P: C-layout -> row-major 16x32 in LDS -> reload as A-fragment
#pragma unroll
    for (int r = 0; r < 8; ++r) {
      smem[widx][r + hi][lr]      = (_Float16)s0[r];
      smem[widx][r + hi][16 + lr] = (_Float16)s1[r];
    }
    asm volatile("s_wait_dscnt 0x0" ::: "memory");
    const _Float16* pr = &smem[widx][lr][(lane >= 16) ? 8 : 0];
    half8 plo = *(const half8*)pr;
    half8 phi = *(const half8*)(pr + 16);
    half16 pf = cat16(plo, phi);

    // O += P * V   (V stored transposed: column d contiguous over keys)
#pragma unroll
    for (int dt = 0; dt < 32; ++dt) {
      half16 bv = load_b_f16(vb + (size_t)(dt * 16 + lr) * NSEQ + kbase, 0, lane);
      o[dt] = wmma_f16(pf, bv, o[dt]);
    }
  }

  // Normalize: softmax denominator and the reference's extra 1/n_kv.
  float invn[8];
#pragma unroll
  for (int r = 0; r < 8; ++r) invn[r] = 1.0f / (l[r] * 2048.0f);

  _Float16* ob = Out + (size_t)b * NSEQ * CDIM;
#pragma unroll
  for (int dt = 0; dt < 32; ++dt)
#pragma unroll
    for (int r = 0; r < 8; ++r) {
      int row = qt * 16 + r + hi;
      int col = dt * 16 + lr;
      ob[(size_t)row * CDIM + col] = (_Float16)(o[dt][r] * invn[r]);
    }
}

// ---------------------------------------------------------------------------
// Kernel 3: final = att(f16)*WoT + bo -> f32 output
// ---------------------------------------------------------------------------
__global__ void proj_out(const _Float16* __restrict__ X, const _Float16* __restrict__ Wt,
                         const float* __restrict__ bias, float* __restrict__ Y) {
  int lane = threadIdx.x & 31;
  int wid  = blockIdx.x * 4 + (threadIdx.x >> 5);
  int rt = wid >> 3, cg = wid & 7;
  int mbase = rt * 16, n0 = cg * 64;
  int lr = lane & 15;

  f32x8 acc[4];
#pragma unroll
  for (int t = 0; t < 4; ++t) acc[t] = zero8();

#pragma unroll
  for (int s = 0; s < 16; ++s) {
    int k0 = 32 * s;
    half16 a = load_a_f16(X, CDIM, mbase + lr, k0, lane);
#pragma unroll
    for (int t = 0; t < 4; ++t) {
      half16 b = load_b_f16(Wt + (size_t)(n0 + t * 16 + lr) * CDIM, k0, lane);
      acc[t] = wmma_f16(a, b, acc[t]);
    }
  }

  int hi = (lane >= 16) ? 8 : 0;
#pragma unroll
  for (int t = 0; t < 4; ++t) {
    int col = n0 + t * 16 + lr;
    float bv = bias[col];
#pragma unroll
    for (int r = 0; r < 8; ++r) {
      int row = mbase + r + hi;
      Y[(size_t)row * CDIM + col] = acc[t][r] + bv;
    }
  }
}

// ---------------------------------------------------------------------------
extern "C" void kernel_launch(void* const* d_in, const int* in_sizes, int n_in,
                              void* d_out, int out_size, void* d_ws, size_t ws_size,
                              hipStream_t stream) {
  const float* query = (const float*)d_in[0];
  const float* key   = (const float*)d_in[1];
  const float* Wq    = (const float*)d_in[2];
  const float* bq    = (const float*)d_in[3];
  const float* Wk    = (const float*)d_in[4];
  const float* bk    = (const float*)d_in[5];
  const float* Wv    = (const float*)d_in[6];
  const float* bv    = (const float*)d_in[7];
  const float* Wo    = (const float*)d_in[8];
  const float* bo    = (const float*)d_in[9];
  float* out = (float*)d_out;

  _Float16* ws = (_Float16*)d_ws;
  const size_t WELEM = (size_t)CDIM * CDIM;            // 256K halfs per weight
  const size_t BND   = (size_t)BSZ * NSEQ * CDIM;      // 32M halfs per tensor
  _Float16* wqT = ws;
  _Float16* wkT = wqT + WELEM;
  _Float16* wvT = wkT + WELEM;
  _Float16* woT = wvT + WELEM;
  _Float16* qh  = woT + WELEM;
  _Float16* kh  = qh + BND;
  _Float16* vth = kh + BND;                            // transposed [B, D, N]
  _Float16* ath = vth + BND;

  prep_weights<<<(CDIM * CDIM) / 256, 256, 0, stream>>>(Wq, Wk, Wv, Wo, wqT, wkT, wvT, woT);

  // (B*N/16) * (C/64) waves = 32768; 4 waves per 128-thread block.
  proj_f32<<<8192, 128, 0, stream>>>(query, wqT, bq, qh, 0);
  proj_f32<<<8192, 128, 0, stream>>>(key,   wkT, bk, kh, 0);
  proj_f32<<<8192, 128, 0, stream>>>(key,   wvT, bv, vth, 1);

  // B * (N/16) waves = 4096; 4 waves per block.
  flash_attn<<<1024, 128, 0, stream>>>(qh, kh, vth, ath);

  proj_out<<<8192, 128, 0, stream>>>(ath, woT, bo, out);
}
